// QuantLlama3Attention_65377992180353
// MI455X (gfx1250) — 8-bit matrix operand order repaired and checked by reading, compiled, not run on hardware
//
#include <hip/hip_runtime.h>

typedef signed char i8;
typedef __attribute__((ext_vector_type(8))) int v8i;
typedef __attribute__((ext_vector_type(4))) int v4i;
typedef __attribute__((address_space(1))) v4i gv4i;
typedef __attribute__((address_space(3))) v4i lv4i;

#if defined(__has_builtin)
#if __has_builtin(__builtin_amdgcn_global_load_async_to_lds_b128) && \
    __has_builtin(__builtin_amdgcn_s_wait_asynccnt)
#define USE_ASYNC 1
#endif
#endif
#ifndef USE_ASYNC
#define USE_ASYNC 0
#endif

constexpr int Bb = 2, Ss = 1024, HID = 4096, NH = 32, NKV = 8, HD = 128, GROUPS = 4;


__device__ __forceinline__ float scale_from(const unsigned* a) {
  float amax = __uint_as_float(*a);
  return amax > 0.f ? amax * (1.0f / 127.0f) : 1.0f;
}

__device__ __forceinline__ int quanti(float x, float inv) {
  int v = (int)rintf(x * inv);
  return v < -128 ? -128 : (v > 127 ? 127 : v);
}

__device__ __forceinline__ v8i load_a_frag(const i8* base, int stride, int lane) {
  int m = lane & 15, half = lane >> 4;
  v8i f;
#pragma unroll
  for (int d = 0; d < 8; ++d) {
    int kb = ((d >> 1) << 4) + (half << 3) + ((d & 1) << 2);
    f[d] = *(const int*)(base + (size_t)m * stride + kb);
  }
  return f;
}

__device__ __forceinline__ v8i load_b_frag(const i8* base, int stride, int lane) {
  int n = lane & 15, half = lane >> 4;
  v8i f;
#pragma unroll
  for (int d = 0; d < 8; ++d) {
    int kb = ((d >> 1) << 4) + (half << 3) + ((d & 1) << 2);
    f[d] = *(const int*)(base + (size_t)n * stride + kb);
  }
  return f;
}

__global__ __launch_bounds__(256) void amax_kernel(const float* __restrict__ x, size_t n,
                                                   unsigned* __restrict__ out) {
  float m = 0.f;
  size_t stride = (size_t)gridDim.x * blockDim.x;
  for (size_t i = (size_t)blockIdx.x * blockDim.x + threadIdx.x; i < n; i += stride)
    m = fmaxf(m, fabsf(x[i]));
#pragma unroll
  for (int o = 16; o > 0; o >>= 1) m = fmaxf(m, __shfl_xor(m, o, 32));
  __shared__ float sm[8];
  if ((threadIdx.x & 31) == 0) sm[threadIdx.x >> 5] = m;
  __syncthreads();
  if (threadIdx.x == 0) {
    float v = sm[0];
#pragma unroll
    for (int w = 1; w < 8; ++w) v = fmaxf(v, sm[w]);
    atomicMax(out, __float_as_uint(v));
  }
}

__global__ __launch_bounds__(256) void quantize_kernel(const float* __restrict__ x,
                                                       i8* __restrict__ q, size_t n,
                                                       const unsigned* amax) {
  float inv = 1.0f / scale_from(amax);
  size_t i = (size_t)blockIdx.x * blockDim.x + threadIdx.x;
  if (i < n) q[i] = (i8)quanti(x[i], inv);
}

__global__ __launch_bounds__(256) void quantize_bhsd_kernel(const float* __restrict__ x,
                                                            i8* __restrict__ q, int H,
                                                            const unsigned* amax) {
  float inv = 1.0f / scale_from(amax);
  size_t n = (size_t)Bb * H * Ss * HD;
  size_t i = (size_t)blockIdx.x * blockDim.x + threadIdx.x;
  if (i >= n) return;
  int d = (int)(i % HD);
  size_t t = i / HD;
  int s = (int)(t % Ss);
  size_t t2 = t / Ss;
  int h = (int)(t2 % H);
  int b = (int)(t2 / H);
  float v = x[((size_t)b * Ss + s) * ((size_t)H * HD) + (size_t)h * HD + d];
  q[i] = (i8)quanti(v, inv);
}

__global__ __launch_bounds__(256) void quantize_vT_kernel(const float* __restrict__ x,
                                                          i8* __restrict__ q,
                                                          const unsigned* amax) {
  float inv = 1.0f / scale_from(amax);
  size_t n = (size_t)Bb * NKV * HD * Ss;
  size_t i = (size_t)blockIdx.x * blockDim.x + threadIdx.x;
  if (i >= n) return;
  int s = (int)(i % Ss);
  size_t t = i / Ss;
  int d = (int)(t % HD);
  size_t t2 = t / HD;
  int kv = (int)(t2 % NKV);
  int b = (int)(t2 / NKV);
  float v = x[((size_t)b * Ss + s) * ((size_t)NKV * HD) + (size_t)kv * HD + d];
  q[i] = (i8)quanti(v, inv);
}

__global__ __launch_bounds__(256) void quantize_ctx_kernel(const float* __restrict__ x,
                                                           i8* __restrict__ q,
                                                           const unsigned* amax) {
  float inv = 1.0f / scale_from(amax);
  size_t n = (size_t)Bb * Ss * NH * HD;
  size_t i = (size_t)blockIdx.x * blockDim.x + threadIdx.x;
  if (i >= n) return;
  int d = (int)(i % HD);
  size_t t = i / HD;
  int h = (int)(t % NH);
  size_t t2 = t / NH;
  int s = (int)(t2 % Ss);
  int b = (int)(t2 / Ss);
  float v = x[(((size_t)b * NH + h) * Ss + s) * HD + d];
  q[i] = (i8)quanti(v, inv);
}

__global__ __launch_bounds__(256) void rope_kernel(float* __restrict__ x,
                                                   const float* __restrict__ cp,
                                                   const float* __restrict__ sp, int H) {
  size_t n = (size_t)Bb * Ss * H * 64;
  size_t i = (size_t)blockIdx.x * blockDim.x + threadIdx.x;
  if (i >= n) return;
  int d = (int)(i % 64);
  size_t t = i / 64;
  int h = (int)(t % H);
  size_t row = t / H;
  float* p = x + row * (size_t)(H * HD) + (size_t)h * HD;
  const float* cr = cp + row * HD;
  const float* sr = sp + row * HD;
  float x1 = p[d], x2 = p[d + 64];
  p[d] = x1 * cr[d] - x2 * sr[d];
  p[d + 64] = x2 * cr[d + 64] + x1 * sr[d + 64];
}

__global__ __launch_bounds__(256) void gemm_i8_kernel(
    const i8* __restrict__ A, const i8* __restrict__ Bm, float* __restrict__ C,
    int M, int N, int K,
    const unsigned* amaxA, float fixedA, const unsigned* amaxB, float extra,
    long long strideA, long long strideB, long long strideC, int gqa) {
  int z = blockIdx.z;
  size_t offA = (size_t)z * (size_t)strideA;
  size_t offC = (size_t)z * (size_t)strideC;
  size_t offB;
  if (gqa) {
    int b = z / NH, h = z % NH;
    offB = ((size_t)b * NKV + (size_t)(h / GROUPS)) * (size_t)strideB;
  } else {
    offB = (size_t)z * (size_t)strideB;
  }

  __shared__ i8 As[2][128 * 64];
  __shared__ i8 Bs[2][128 * 64];

  int tid = threadIdx.x;
  int wave = tid >> 5, lane = tid & 31;
  int wm = wave & 1, wn = wave >> 1;
  int m0 = blockIdx.x * 128;
  int n0 = blockIdx.y * 128;

  const i8* Ag = A + offA;
  const i8* Bg = Bm + offB;

  int r0 = tid >> 2;
  int s0 = (tid & 3) << 4;

  auto issue = [&](int k0, int buf) {
    const i8* ga0 = Ag + (size_t)(m0 + r0) * K + k0 + s0;
    const i8* ga1 = Ag + (size_t)(m0 + r0 + 64) * K + k0 + s0;
    const i8* gb0 = Bg + (size_t)(n0 + r0) * K + k0 + s0;
    const i8* gb1 = Bg + (size_t)(n0 + r0 + 64) * K + k0 + s0;
    i8* la = &As[buf][0];
    i8* lb = &Bs[buf][0];
#if USE_ASYNC
    __builtin_amdgcn_global_load_async_to_lds_b128(
        (gv4i*)ga0, (lv4i*)(la + r0 * 64 + s0), 0, 0);
    __builtin_amdgcn_global_load_async_to_lds_b128(
        (gv4i*)ga1, (lv4i*)(la + (r0 + 64) * 64 + s0), 0, 0);
    __builtin_amdgcn_global_load_async_to_lds_b128(
        (gv4i*)gb0, (lv4i*)(lb + r0 * 64 + s0), 0, 0);
    __builtin_amdgcn_global_load_async_to_lds_b128(
        (gv4i*)gb1, (lv4i*)(lb + (r0 + 64) * 64 + s0), 0, 0);
#else
    int4 a0 = *(const int4*)ga0;
    int4 a1 = *(const int4*)ga1;
    int4 b0 = *(const int4*)gb0;
    int4 b1 = *(const int4*)gb1;
    *(int4*)(la + r0 * 64 + s0) = a0;
    *(int4*)(la + (r0 + 64) * 64 + s0) = a1;
    *(int4*)(lb + r0 * 64 + s0) = b0;
    *(int4*)(lb + (r0 + 64) * 64 + s0) = b1;
#endif
  };

  v8i acc[4][2] = {};
  int nk = K >> 6;
  issue(0, 0);

  for (int kt = 0; kt < nk; ++kt) {
    int buf = kt & 1;
#if USE_ASYNC
    __builtin_amdgcn_s_wait_asynccnt(0);
#endif
    __syncthreads();
    if (kt + 1 < nk) issue((kt + 1) << 6, buf ^ 1);

    v8i af[4];
#pragma unroll
    for (int i2 = 0; i2 < 4; ++i2)
      af[i2] = load_a_frag(&As[buf][(size_t)(wm * 64 + i2 * 16) * 64], 64, lane);
#pragma unroll
    for (int j = 0; j < 2; ++j) {
      v8i bf = load_b_frag(&Bs[buf][(size_t)(wn * 32 + j * 16) * 64], 64, lane);
#pragma unroll
      for (int i2 = 0; i2 < 4; ++i2) {
        acc[i2][j] = __builtin_amdgcn_wmma_i32_16x16x64_iu8(
            true, af[i2], true, bf, acc[i2][j], false, false);
      }
    }
  }

  float sA = amaxA ? scale_from(amaxA) : fixedA;
  float sB = scale_from(amaxB);
  float s = sA * sB * extra;
  int half = lane >> 4, nl = lane & 15;
#pragma unroll
  for (int i2 = 0; i2 < 4; ++i2)
#pragma unroll
    for (int j = 0; j < 2; ++j)
#pragma unroll
      for (int r = 0; r < 8; ++r) {
        int row = m0 + wm * 64 + i2 * 16 + r + half * 8;
        int col = n0 + wn * 32 + j * 16 + nl;
        C[offC + (size_t)row * N + col] = s * (float)acc[i2][j][r];
      }
}

__global__ __launch_bounds__(256) void attn_kernel(const i8* __restrict__ q8,
                                                   const i8* __restrict__ k8,
                                                   float* __restrict__ probs,
                                                   i8* __restrict__ p8,
                                                   const unsigned* amaxQ,
                                                   const unsigned* amaxK) {
  __shared__ float sc[16 * Ss];

  int qt = blockIdx.x, h = blockIdx.y, b = blockIdx.z;
  int kv = h / GROUPS;
  int tid = threadIdx.x, wave = tid >> 5, lane = tid & 31;
  int half = lane >> 4, nl = lane & 15;

  const i8* qb = q8 + (((size_t)b * NH + h) * Ss + (size_t)qt * 16) * HD;
  const i8* kb = k8 + ((size_t)b * NKV + kv) * (size_t)Ss * HD;

  float scale = scale_from(amaxQ) * scale_from(amaxK) * 0.08838834764831845f;

  for (int nt = 0; nt < 8; ++nt) {
    int kb0 = wave * 128 + nt * 16;
    v8i acc = {};
#pragma unroll
    for (int c = 0; c < 2; ++c) {
      v8i af = load_a_frag(qb + c * 64, HD, lane);
      v8i bf = load_b_frag(kb + (size_t)kb0 * HD + c * 64, HD, lane);
      acc = __builtin_amdgcn_wmma_i32_16x16x64_iu8(true, af, true, bf, acc, false, false);
    }
#pragma unroll
    for (int r = 0; r < 8; ++r) {
      int row = r + half * 8;
      int col = kb0 + nl;
      int qpos = qt * 16 + row;
      float v = scale * (float)acc[r];
      sc[row * Ss + col] = (col <= qpos) ? v : -3.402823466e38f;
    }
  }
  __syncthreads();

  for (int rr = 0; rr < 2; ++rr) {
    int row = wave + rr * 8;
    float* srow = sc + (size_t)row * Ss;
    float m = -3.402823466e38f;
    for (int j = lane; j < Ss; j += 32) m = fmaxf(m, srow[j]);
#pragma unroll
    for (int o = 16; o > 0; o >>= 1) m = fmaxf(m, __shfl_xor(m, o, 32));
    float sum = 0.f;
    for (int j = lane; j < Ss; j += 32) {
      float e = __expf(srow[j] - m);
      srow[j] = e;
      sum += e;
    }
#pragma unroll
    for (int o = 16; o > 0; o >>= 1) sum += __shfl_xor(sum, o, 32);
    float inv = 1.0f / sum;
    size_t orow = (((size_t)b * NH + h) * Ss + (size_t)qt * 16 + row) * Ss;
    for (int j = lane; j < Ss; j += 32) {
      float p = srow[j] * inv;
      probs[orow + j] = p;
      p8[orow + j] = (i8)quanti(p, 127.0f);
    }
  }
}

extern "C" void kernel_launch(void* const* d_in, const int* in_sizes, int n_in,
                              void* d_out, int out_size, void* d_ws, size_t ws_size,
                              hipStream_t stream) {
  const float* hs = (const float*)d_in[0];
  const float* cosp = (const float*)d_in[1];
  const float* sinp = (const float*)d_in[2];
  const float* wq = (const float*)d_in[4];
  const float* wk = (const float*)d_in[5];
  const float* wv = (const float*)d_in[6];
  const float* wo = (const float*)d_in[7];

  char* w = (char*)d_ws;
  size_t o = 0;
  auto take = [&](size_t sz) -> void* {
    void* p = w + o;
    o = (o + sz + 255) & ~(size_t)255;
    return p;
  };
  unsigned* sc = (unsigned*)take(64);
  i8* hs8 = (i8*)take((size_t)Bb * Ss * HID);
  i8* wq8 = (i8*)take((size_t)NH * HD * HID);
  i8* wk8 = (i8*)take((size_t)NKV * HD * HID);
  i8* wv8 = (i8*)take((size_t)NKV * HD * HID);
  i8* wo8 = (i8*)take((size_t)HID * NH * HD);
  float* qf = (float*)take((size_t)Bb * Ss * NH * HD * 4);
  float* kf = (float*)take((size_t)Bb * Ss * NKV * HD * 4);
  float* vf = (float*)take((size_t)Bb * Ss * NKV * HD * 4);
  i8* q8 = (i8*)take((size_t)Bb * NH * Ss * HD);
  i8* k8 = (i8*)take((size_t)Bb * NKV * Ss * HD);
  i8* v8t = (i8*)take((size_t)Bb * NKV * HD * Ss);
  i8* p8 = (i8*)take((size_t)Bb * NH * Ss * Ss);
  float* ctxf = (float*)take((size_t)Bb * NH * Ss * HD * 4);
  i8* ctx8 = (i8*)take((size_t)Bb * Ss * NH * HD);

  float* out_attn = (float*)d_out;
  float* out_probs = out_attn + (size_t)Bb * Ss * HID;

  (void)hipMemsetAsync(sc, 0, 64, stream);

  auto nb = [](size_t n) { return dim3((unsigned)((n + 255) / 256)); };
  auto nbcap = [](size_t n) {
    size_t bl = (n + 255) / 256;
    if (bl > 2048) bl = 2048;
    return dim3((unsigned)bl);
  };

  const size_t n_hs = (size_t)Bb * Ss * HID;
  const size_t n_wq = (size_t)NH * HD * HID;
  const size_t n_wk = (size_t)NKV * HD * HID;
  const size_t n_q = (size_t)Bb * Ss * NH * HD;
  const size_t n_k = (size_t)Bb * Ss * NKV * HD;

  amax_kernel<<<nbcap(n_hs), 256, 0, stream>>>(hs, n_hs, sc + 0);
  amax_kernel<<<nbcap(n_wq), 256, 0, stream>>>(wq, n_wq, sc + 1);
  amax_kernel<<<nbcap(n_wk), 256, 0, stream>>>(wk, n_wk, sc + 2);
  amax_kernel<<<nbcap(n_wk), 256, 0, stream>>>(wv, n_wk, sc + 3);
  amax_kernel<<<nbcap(n_wq), 256, 0, stream>>>(wo, n_wq, sc + 4);

  quantize_kernel<<<nb(n_hs), 256, 0, stream>>>(hs, hs8, n_hs, sc + 0);
  quantize_kernel<<<nb(n_wq), 256, 0, stream>>>(wq, wq8, n_wq, sc + 1);
  quantize_kernel<<<nb(n_wk), 256, 0, stream>>>(wk, wk8, n_wk, sc + 2);
  quantize_kernel<<<nb(n_wk), 256, 0, stream>>>(wv, wv8, n_wk, sc + 3);
  quantize_kernel<<<nb(n_wq), 256, 0, stream>>>(wo, wo8, n_wq, sc + 4);

  gemm_i8_kernel<<<dim3(16, 32, 1), 256, 0, stream>>>(hs8, wq8, qf, 2048, 4096, 4096,
                                                      sc + 0, 0.f, sc + 1, 1.0f, 0, 0, 0, 0);
  gemm_i8_kernel<<<dim3(16, 8, 1), 256, 0, stream>>>(hs8, wk8, kf, 2048, 1024, 4096,
                                                     sc + 0, 0.f, sc + 2, 1.0f, 0, 0, 0, 0);
  gemm_i8_kernel<<<dim3(16, 8, 1), 256, 0, stream>>>(hs8, wv8, vf, 2048, 1024, 4096,
                                                     sc + 0, 0.f, sc + 3, 1.0f, 0, 0, 0, 0);

  rope_kernel<<<nb((size_t)Bb * Ss * NH * 64), 256, 0, stream>>>(qf, cosp, sinp, NH);
  rope_kernel<<<nb((size_t)Bb * Ss * NKV * 64), 256, 0, stream>>>(kf, cosp, sinp, NKV);

  amax_kernel<<<nbcap(n_q), 256, 0, stream>>>(qf, n_q, sc + 5);
  amax_kernel<<<nbcap(n_k), 256, 0, stream>>>(kf, n_k, sc + 6);
  amax_kernel<<<nbcap(n_k), 256, 0, stream>>>(vf, n_k, sc + 7);
  quantize_bhsd_kernel<<<nb(n_q), 256, 0, stream>>>(qf, q8, NH, sc + 5);
  quantize_bhsd_kernel<<<nb(n_k), 256, 0, stream>>>(kf, k8, NKV, sc + 6);
  quantize_vT_kernel<<<nb(n_k), 256, 0, stream>>>(vf, v8t, sc + 7);

  attn_kernel<<<dim3(Ss / 16, NH, Bb), 256, 0, stream>>>(q8, k8, out_probs, p8,
                                                         sc + 5, sc + 6);

  gemm_i8_kernel<<<dim3(8, 1, Bb * NH), 256, 0, stream>>>(
      p8, v8t, ctxf, 1024, 128, 1024, nullptr, 1.0f / 127.0f, sc + 7, 1.0f,
      (long long)Ss * Ss, (long long)HD * Ss, (long long)Ss * HD, 1);

  amax_kernel<<<nbcap(n_q), 256, 0, stream>>>(ctxf, n_q, sc + 8);
  quantize_ctx_kernel<<<nb(n_q), 256, 0, stream>>>(ctxf, ctx8, sc + 8);
  gemm_i8_kernel<<<dim3(16, 32, 1), 256, 0, stream>>>(ctx8, wo8, out_attn, 2048, 4096, 4096,
                                                      sc + 8, 0.f, sc + 4, 1.0f, 0, 0, 0, 0);
}
